// DualPathTransformerLayer_54752243089799
// MI455X (gfx1250) — compile-verified, matrix operand order checked by reading, not run on hardware
//
#include <hip/hip_runtime.h>
#include <hip/hip_bf16.h>

// ---------------------------------------------------------------------------
// DualPathTransformerLayer for MI455X (gfx1250, wave32, WMMA).
// - All matmuls on v_wmma_f32_16x16x32_f16, f32 accumulation.
// - Weights pre-packed into ISA B-fragment layout (pure b128 loads + wmma,
//   software pipelined, 16x64 tile per wave).
// - V pre-transposed to [b,h,dv][key] so attention V-fragments are b128 loads.
// - Mem-path streaming softmax in row-major domain via per-wave LDS S spill;
//   all softmax state kept in the log2 domain (raw v_exp_f32, no ln2 muls).
// ---------------------------------------------------------------------------

typedef __attribute__((ext_vector_type(16))) _Float16 v16h;
typedef __attribute__((ext_vector_type(8)))  float    v8f;

#define INVC  0.99950037f        // 1/sqrt(1+1e-3)
#define LOG2E 1.4426950408889634f

__device__ __forceinline__ v8f wmma16(v16h a, v16h b, v8f c) {
  return __builtin_amdgcn_wmma_f32_16x16x32_f16(
      false, a, false, b, (short)0, c, false, false);
}

__device__ __forceinline__ float fast_exp2(float x) {
#if __has_builtin(__builtin_amdgcn_exp2f)
  return __builtin_amdgcn_exp2f(x);
#else
  return exp2f(x);
#endif
}
__device__ __forceinline__ float fast_rcp(float x) {
#if __has_builtin(__builtin_amdgcn_rcpf)
  return __builtin_amdgcn_rcpf(x);
#else
  return 1.f / x;
#endif
}

// ---------------------------------------------------------------------------
// Pack W (K x Nc, f32 row-major) into f16 B-fragment layout:
//   Wp[((nt*KT + kt)*32 + lane)*16 + j], KT = K/32.
// ---------------------------------------------------------------------------
__global__ __launch_bounds__(256) void pack_w_kernel(
    const float* __restrict__ W, _Float16* __restrict__ Wp, int K, int Nc)
{
  const int tid = (int)(blockIdx.x * blockDim.x + threadIdx.x);
  const int KT = K >> 5;
  const int per_nt = KT * 32;
  const int nt = tid / per_nt;
  if (nt >= (Nc >> 4)) return;
  const int rem  = tid - nt * per_nt;
  const int kt   = rem >> 5;
  const int lane = rem & 31;
  const int l15  = lane & 15;
  const int koff = (lane >> 4) << 3;
  const int col  = nt * 16 + l15;
  _Float16* dst = Wp + (size_t)tid * 16;
#pragma unroll
  for (int j = 0; j < 8; ++j) {
    dst[j]     = (_Float16)W[(size_t)(kt * 32 + koff + j) * Nc + col];
    dst[j + 8] = (_Float16)W[(size_t)(kt * 32 + 16 + koff + j) * Nc + col];
  }
}

// ---------------------------------------------------------------------------
// Fused GEMM + BN (+ residual, + ReLU), packed-f16 weights, pipelined.
// ---------------------------------------------------------------------------
template<typename TA, typename TR, bool RELU, bool RESID, bool OUT16>
__global__ __launch_bounds__(256) void gemm_bn_kernel(
    const TA* __restrict__ A, const _Float16* __restrict__ Wp,
    const float* __restrict__ bn, const TR* __restrict__ resid,
    _Float16* __restrict__ out16, float* __restrict__ out32,
    int M, int K, int Nc)
{
  const int wave = (int)((blockIdx.x * blockDim.x + threadIdx.x) >> 5);
  const int lane = threadIdx.x & 31;
  const int KT = K >> 5;
  const int ngroups = Nc >> 6;
  const int mt = wave / ngroups;
  const int ng = wave - mt * ngroups;
  if (mt >= (M >> 4)) return;

  const int half = lane >> 4, l15 = lane & 15, koff = half << 3;
  const size_t arow = (size_t)(mt * 16 + l15) * (size_t)K;

  const _Float16* wb0 = Wp + ((size_t)((ng * 4 + 0) * KT) * 32 + lane) * 16;
  const _Float16* wb1 = Wp + ((size_t)((ng * 4 + 1) * KT) * 32 + lane) * 16;
  const _Float16* wb2 = Wp + ((size_t)((ng * 4 + 2) * KT) * 32 + lane) * 16;
  const _Float16* wb3 = Wp + ((size_t)((ng * 4 + 3) * KT) * 32 + lane) * 16;

  auto loadA = [&](int kt) -> v16h {
    v16h a;
#pragma unroll
    for (int j = 0; j < 8; ++j) {
      a[j]     = (_Float16)A[arow + (size_t)(kt * 32 + koff + j)];
      a[j + 8] = (_Float16)A[arow + (size_t)(kt * 32 + 16 + koff + j)];
    }
    return a;
  };
  auto loadB = [&](const _Float16* base, int kt) -> v16h {
    return *(const v16h*)(base + (size_t)kt * 512);
  };

  v8f acc0 = {}, acc1 = {}, acc2 = {}, acc3 = {};
  v16h a  = loadA(0);
  v16h b0 = loadB(wb0, 0), b1 = loadB(wb1, 0);
  v16h b2 = loadB(wb2, 0), b3 = loadB(wb3, 0);

  for (int kt = 0; kt < KT - 1; ++kt) {
    __builtin_prefetch(&A[arow + (size_t)((kt + 2) * 32)], 0, 3);
    v16h an  = loadA(kt + 1);
    v16h bn0 = loadB(wb0, kt + 1), bn1 = loadB(wb1, kt + 1);
    v16h bn2 = loadB(wb2, kt + 1), bn3 = loadB(wb3, kt + 1);
    acc0 = wmma16(a, b0, acc0);
    acc1 = wmma16(a, b1, acc1);
    acc2 = wmma16(a, b2, acc2);
    acc3 = wmma16(a, b3, acc3);
    a = an; b0 = bn0; b1 = bn1; b2 = bn2; b3 = bn3;
  }
  acc0 = wmma16(a, b0, acc0);
  acc1 = wmma16(a, b1, acc1);
  acc2 = wmma16(a, b2, acc2);
  acc3 = wmma16(a, b3, acc3);

#pragma unroll
  for (int i = 0; i < 4; ++i) {
    const int col = (ng * 4 + i) * 16 + l15;
    const float g  = bn[col] * INVC;
    const float be = bn[Nc + col];
    const v8f acc = (i == 0) ? acc0 : (i == 1) ? acc1 : (i == 2) ? acc2 : acc3;
#pragma unroll
    for (int e = 0; e < 8; ++e) {
      const size_t r = (size_t)(mt * 16 + e + half * 8);
      float v = acc[e] * g + be;
      if (RESID) v += (float)resid[r * (size_t)Nc + col];
      if (RELU)  v = fmaxf(v, 0.f);
      if (OUT16) out16[r * (size_t)Nc + col] = (_Float16)v;
      else       out32[r * (size_t)Nc + col] = v;
    }
  }
}

// ---------------------------------------------------------------------------
// Transpose V into vT[((b*8+h)*32 + dv)*4224 + key], key 0..4095 = pixel
// tokens, 4096..4223 = memory tokens.
// ---------------------------------------------------------------------------
__global__ __launch_bounds__(256) void transpose_v_kernel(
    const _Float16* __restrict__ mqkv, const _Float16* __restrict__ pqkv,
    _Float16* __restrict__ vT)
{
  const int tid = (int)(blockIdx.x * blockDim.x + threadIdx.x);
  const int kg = tid % 528;
  const int dv = (tid / 528) & 31;
  const int h  = (tid / (528 * 32)) & 7;
  const int b  =  tid / (528 * 32 * 8);
  const int k0 = kg * 8;
  _Float16 tmp[8];
  if (k0 < 4096) {
    const _Float16* src = pqkv + ((size_t)(b * 4096 + k0)) * 512 + 256 + h * 32 + dv;
#pragma unroll
    for (int j = 0; j < 8; ++j) tmp[j] = src[(size_t)j * 512];
  } else {
    const _Float16* src = mqkv + ((size_t)(b * 128 + (k0 - 4096))) * 512 + 256 + h * 32 + dv;
#pragma unroll
    for (int j = 0; j < 8; ++j) tmp[j] = src[(size_t)j * 512];
  }
  _Float16* dst = vT + (((size_t)(b * 8 + h)) * 32 + dv) * 4224 + k0;
#pragma unroll
  for (int j = 0; j < 8; ++j) dst[j] = tmp[j];
}

// ---------------------------------------------------------------------------
// Memory-path attention: per (b,h), 128 queries over 4224 keys, streaming
// softmax (log2 domain) in row-major form via per-wave LDS S spill.
// ---------------------------------------------------------------------------
__global__ __launch_bounds__(256) void mem_attn_kernel(
    const _Float16* __restrict__ mqkv, const _Float16* __restrict__ pqkv,
    const _Float16* __restrict__ vT,
    const float* __restrict__ bn_sim, const float* __restrict__ bn_ret,
    _Float16* __restrict__ ret)
{
  __shared__ float smem[8 * 512];              // per-wave 16x32 f32 S tile
  const int wv   = threadIdx.x >> 5;
  const int lane = threadIdx.x & 31;
  const int b    = blockIdx.x >> 3;
  const int h    = blockIdx.x & 7;
  const int qbase = wv * 16;
  const int half = lane >> 4, l15 = lane & 15, koff = half << 3;
  float*     sw = smem + wv * 512;             // f32 view (S tile)
  _Float16*  ph = (_Float16*)sw;               // f16 view (P tile, same bytes)

  v16h qf = {};
  {
    const _Float16* qp = mqkv + ((size_t)(b * 128 + qbase + l15)) * 512 + h * 16;
#pragma unroll
    for (int j = 0; j < 8; ++j) qf[j] = qp[koff + j];
  }

  // BN scale/bias folded with log2(e): softmax computed in the log2 domain.
  const float gs = bn_sim[h] * INVC * LOG2E, bb = bn_sim[8 + h] * LOG2E;
  const _Float16* vreg = vT + ((size_t)(b * 8 + h)) * 32 * 4224;
  float mrow = -__builtin_inff();
  float srow = 0.f;
  v8f o0 = {}, o1 = {};

  auto chunk = [&](const _Float16* kreg, int kl, int vk) {
    const _Float16* kp0 = kreg + (size_t)(kl + l15) * 512 + 128 + h * 16 + koff;
    const _Float16* kp1 = kp0 + (size_t)16 * 512;
    __builtin_prefetch(kp0 + (size_t)32 * 512, 0, 3);
    v16h kf0 = {}, kf1 = {};
#pragma unroll
    for (int j = 0; j < 8; ++j) { kf0[j] = kp0[j]; kf1[j] = kp1[j]; }
    v8f s0 = {}, s1 = {};
    s0 = wmma16(qf, kf0, s0);
    s1 = wmma16(qf, kf1, s1);

    const _Float16* vb0 = vreg + (size_t)l15 * 4224 + vk;
    const _Float16* vb1 = vreg + (size_t)(16 + l15) * 4224 + vk;
    v16h vf0, vf1;
#pragma unroll
    for (int j = 0; j < 8; ++j) {
      vf0[j]     = vb0[koff + j];
      vf0[j + 8] = vb0[16 + koff + j];
      vf1[j]     = vb1[koff + j];
      vf1[j + 8] = vb1[16 + koff + j];
    }

    // Spill S (BN+log2e folded) to LDS row-major f32.
#pragma unroll
    for (int e = 0; e < 8; ++e) {
      sw[(e + 8 * half) * 32 + l15]      = s0[e] * gs + bb;
      sw[(e + 8 * half) * 32 + 16 + l15] = s1[e] * gs + bb;
    }
    // Row-domain softmax: lane owns row l15, columns half*16..+15.
    const float4* rp4 = (const float4*)(sw + l15 * 32 + half * 16);
    float av[16];
#pragma unroll
    for (int q = 0; q < 4; ++q) {
      const float4 r4 = rp4[q];
      av[q * 4 + 0] = r4.x; av[q * 4 + 1] = r4.y;
      av[q * 4 + 2] = r4.z; av[q * 4 + 3] = r4.w;
    }
    float mx = av[0];
#pragma unroll
    for (int i = 1; i < 16; ++i) mx = fmaxf(mx, av[i]);
    mx = fmaxf(mx, __shfl_xor(mx, 16, 32));
    const float nm  = fmaxf(mrow, mx);
    const float fac = fast_exp2(mrow - nm);
    mrow = nm;
    v16h pv;
    float rs = 0.f;
#pragma unroll
    for (int i = 0; i < 16; ++i) {
      const float pe = fast_exp2(av[i] - nm);
      rs += pe;
      pv[i] = (_Float16)pe;
    }
    rs += __shfl_xor(rs, 16, 32);
    srow = srow * fac + rs;
    *(v16h*)(ph + l15 * 32 + half * 16) = pv;

#pragma unroll
    for (int e = 0; e < 8; ++e) {
      const float f = __shfl(fac, e + 8 * half, 16);
      o0[e] *= f; o1[e] *= f;
    }
    v16h pf;
#pragma unroll
    for (int j = 0; j < 8; ++j) {
      pf[j]     = ph[l15 * 32 + koff + j];
      pf[j + 8] = ph[l15 * 32 + 16 + koff + j];
    }
    o0 = wmma16(pf, vf0, o0);
    o1 = wmma16(pf, vf1, o1);
  };

  const _Float16* preg = pqkv + (size_t)b * 4096 * 512;
  const _Float16* mreg = mqkv + (size_t)b * 128 * 512;
  for (int c = 0; c < 4096; c += 32) chunk(preg, c, c);
  for (int c = 0; c < 128;  c += 32) chunk(mreg, c, 4096 + c);

  const float inv = fast_rcp(srow);
#pragma unroll
  for (int t = 0; t < 2; ++t) {
    const int dv = t * 16 + l15;
    const float g  = bn_ret[h * 32 + dv] * INVC;
    const float be = bn_ret[256 + h * 32 + dv];
#pragma unroll
    for (int e = 0; e < 8; ++e) {
      const float iv = __shfl(inv, e + 8 * half, 16);
      const int r = qbase + e + half * 8;
      const float ov = (t == 0) ? o0[e] : o1[e];
      float v = (ov * iv) * g + be;
      v = fmaxf(v, 0.f);
      ret[((size_t)(b * 128 + r)) * 256 + h * 32 + dv] = (_Float16)v;
    }
  }
}

// ---------------------------------------------------------------------------
// Pixel-path attention: 4096 queries x 128 memory keys, sigmoid gating
// (rcp(1+exp2(-a)) with log2e folded into BN), pret = sig(S)*V, BN+ReLU.
// ---------------------------------------------------------------------------
__global__ __launch_bounds__(256) void pix_attn_kernel(
    const _Float16* __restrict__ mqkv, const _Float16* __restrict__ pqkv,
    const _Float16* __restrict__ vT,
    const float* __restrict__ bn_sim, const float* __restrict__ bn_ret,
    _Float16* __restrict__ pret)
{
  __shared__ _Float16 plds[8 * 16 * 32];
  const int gw   = (int)(blockIdx.x * 8 + (threadIdx.x >> 5));
  const int lane = threadIdx.x & 31;
  const int b  = gw >> 11;
  const int h  = (gw >> 8) & 7;
  const int qt = gw & 255;
  const int qbase = qt * 16;
  const int half = lane >> 4, l15 = lane & 15, koff = half << 3;
  _Float16* pw = plds + (threadIdx.x >> 5) * 16 * 32;

  v16h qf = {};
  {
    const _Float16* qp = pqkv + ((size_t)(b * 4096 + qbase + l15)) * 512 + h * 16;
#pragma unroll
    for (int j = 0; j < 8; ++j) qf[j] = qp[koff + j];
  }

  const float gs = bn_sim[h] * INVC * LOG2E, bb = bn_sim[8 + h] * LOG2E;
  const _Float16* mreg = mqkv + (size_t)b * 128 * 512;
  const _Float16* vreg = vT + ((size_t)(b * 8 + h)) * 32 * 4224 + 4096;
  v8f o0 = {}, o1 = {};

  for (int c = 0; c < 128; c += 32) {
    const _Float16* kp0 = mreg + (size_t)(c + l15) * 512 + 128 + h * 16 + koff;
    const _Float16* kp1 = kp0 + (size_t)16 * 512;
    v16h kf0 = {}, kf1 = {};
#pragma unroll
    for (int j = 0; j < 8; ++j) { kf0[j] = kp0[j]; kf1[j] = kp1[j]; }
    v8f s0 = {}, s1 = {};
    s0 = wmma16(qf, kf0, s0);
    s1 = wmma16(qf, kf1, s1);

    const _Float16* vb0 = vreg + (size_t)l15 * 4224 + c;
    const _Float16* vb1 = vreg + (size_t)(16 + l15) * 4224 + c;
    v16h vf0, vf1;
#pragma unroll
    for (int j = 0; j < 8; ++j) {
      vf0[j]     = vb0[koff + j];
      vf0[j + 8] = vb0[16 + koff + j];
      vf1[j]     = vb1[koff + j];
      vf1[j + 8] = vb1[16 + koff + j];
    }

#pragma unroll
    for (int e = 0; e < 8; ++e) {
      const float p0 = fast_rcp(1.f + fast_exp2(-(s0[e] * gs + bb)));
      const float p1 = fast_rcp(1.f + fast_exp2(-(s1[e] * gs + bb)));
      const int r = e + half * 8;
      pw[r * 32 + l15]      = (_Float16)p0;
      pw[r * 32 + 16 + l15] = (_Float16)p1;
    }
    v16h pf;
#pragma unroll
    for (int j = 0; j < 8; ++j) {
      pf[j]     = pw[l15 * 32 + koff + j];
      pf[j + 8] = pw[l15 * 32 + 16 + koff + j];
    }
    o0 = wmma16(pf, vf0, o0);
    o1 = wmma16(pf, vf1, o1);
  }

#pragma unroll
  for (int t = 0; t < 2; ++t) {
    const int dv = t * 16 + l15;
    const float g  = bn_ret[h * 32 + dv] * INVC;
    const float be = bn_ret[256 + h * 32 + dv];
#pragma unroll
    for (int e = 0; e < 8; ++e) {
      const int r = qbase + e + half * 8;
      const float ov = (t == 0) ? o0[e] : o1[e];
      float v = ov * g + be;
      v = fmaxf(v, 0.f);
      pret[((size_t)(b * 4096 + r)) * 256 + h * 32 + dv] = (_Float16)v;
    }
  }
}

// ---------------------------------------------------------------------------
extern "C" void kernel_launch(void* const* d_in, const int* in_sizes, int n_in,
                              void* d_out, int out_size, void* d_ws, size_t ws_size,
                              hipStream_t stream) {
  (void)in_sizes; (void)n_in; (void)out_size; (void)ws_size;
  const float* pixel    = (const float*)d_in[0];
  const float* memin    = (const float*)d_in[1];
  const float* W_mem1   = (const float*)d_in[2];
  const float* bn_mem1  = (const float*)d_in[3];
  const float* W_pix1   = (const float*)d_in[4];
  const float* bn_pix1  = (const float*)d_in[5];
  const float* W_mqkv   = (const float*)d_in[6];
  const float* bn_mqkv  = (const float*)d_in[7];
  const float* W_pqkv   = (const float*)d_in[8];
  const float* bn_pqkv  = (const float*)d_in[9];
  const float* bn_msim  = (const float*)d_in[10];
  const float* bn_mret  = (const float*)d_in[11];
  const float* bn_psim  = (const float*)d_in[12];
  const float* bn_pret  = (const float*)d_in[13];
  const float* W_mem3   = (const float*)d_in[14];
  const float* bn_mem3  = (const float*)d_in[15];
  const float* W_pix3   = (const float*)d_in[16];
  const float* bn_pix3  = (const float*)d_in[17];
  const float* W_ffn1   = (const float*)d_in[18];
  const float* bn_ffn1  = (const float*)d_in[19];
  const float* W_ffn2   = (const float*)d_in[20];
  const float* bn_ffn2  = (const float*)d_in[21];

  float* pix_out = (float*)d_out;
  float* mem_out = (float*)d_out + (size_t)8 * 4096 * 128;

  size_t off = 0;
  char* wsb = (char*)d_ws;
  auto take = [&](size_t n) { _Float16* p = (_Float16*)(wsb + off);
                              off += n * sizeof(_Float16); return p; };
  _Float16* memb  = take((size_t)1024 * 256);
  _Float16* pixb  = take((size_t)32768 * 256);
  _Float16* mqkvb = take((size_t)1024 * 512);
  _Float16* pqkvb = take((size_t)32768 * 512);
  _Float16* retb  = take((size_t)1024 * 256);
  _Float16* memo1 = take((size_t)1024 * 128);
  _Float16* ffnb  = take((size_t)1024 * 2048);
  _Float16* pretb = take((size_t)32768 * 256);
  _Float16* vTb   = take((size_t)8 * 8 * 32 * 4224);
  _Float16* wp_mem1 = take((size_t)128 * 256);
  _Float16* wp_pix1 = take((size_t)128 * 256);
  _Float16* wp_mqkv = take((size_t)256 * 512);
  _Float16* wp_pqkv = take((size_t)256 * 512);
  _Float16* wp_mem3 = take((size_t)256 * 128);
  _Float16* wp_pix3 = take((size_t)256 * 128);
  _Float16* wp_ffn1 = take((size_t)128 * 2048);
  _Float16* wp_ffn2 = take((size_t)2048 * 128);

  pack_w_kernel<<<8,  256, 0, stream>>>(W_mem1, wp_mem1, 128, 256);
  pack_w_kernel<<<8,  256, 0, stream>>>(W_pix1, wp_pix1, 128, 256);
  pack_w_kernel<<<32, 256, 0, stream>>>(W_mqkv, wp_mqkv, 256, 512);
  pack_w_kernel<<<32, 256, 0, stream>>>(W_pqkv, wp_pqkv, 256, 512);
  pack_w_kernel<<<8,  256, 0, stream>>>(W_mem3, wp_mem3, 256, 128);
  pack_w_kernel<<<8,  256, 0, stream>>>(W_pix3, wp_pix3, 256, 128);
  pack_w_kernel<<<64, 256, 0, stream>>>(W_ffn1, wp_ffn1, 128, 2048);
  pack_w_kernel<<<64, 256, 0, stream>>>(W_ffn2, wp_ffn2, 2048, 128);

  const float* nf = nullptr;

  gemm_bn_kernel<float, float, true, false, true><<<32, 256, 0, stream>>>(
      memin, wp_mem1, bn_mem1, nf, memb, nullptr, 1024, 128, 256);
  gemm_bn_kernel<float, float, true, false, true><<<1024, 256, 0, stream>>>(
      pixel, wp_pix1, bn_pix1, nf, pixb, nullptr, 32768, 128, 256);
  gemm_bn_kernel<_Float16, float, false, false, true><<<64, 256, 0, stream>>>(
      memb, wp_mqkv, bn_mqkv, nf, mqkvb, nullptr, 1024, 256, 512);
  gemm_bn_kernel<_Float16, float, false, false, true><<<2048, 256, 0, stream>>>(
      pixb, wp_pqkv, bn_pqkv, nf, pqkvb, nullptr, 32768, 256, 512);

  transpose_v_kernel<<<4224, 256, 0, stream>>>(mqkvb, pqkvb, vTb);

  mem_attn_kernel<<<64, 256, 0, stream>>>(mqkvb, pqkvb, vTb, bn_msim, bn_mret, retb);
  pix_attn_kernel<<<2048, 256, 0, stream>>>(mqkvb, pqkvb, vTb, bn_psim, bn_pret, pretb);

  gemm_bn_kernel<_Float16, float, true, true, true><<<16, 256, 0, stream>>>(
      retb, wp_mem3, bn_mem3, memin, memo1, nullptr, 1024, 256, 128);
  gemm_bn_kernel<_Float16, float, true, false, true><<<256, 256, 0, stream>>>(
      memo1, wp_ffn1, bn_ffn1, nf, ffnb, nullptr, 1024, 128, 2048);
  gemm_bn_kernel<_Float16, _Float16, true, true, false><<<16, 256, 0, stream>>>(
      ffnb, wp_ffn2, bn_ffn2, memo1, nullptr, mem_out, 1024, 2048, 128);
  gemm_bn_kernel<_Float16, float, true, true, false><<<512, 256, 0, stream>>>(
      pretb, wp_pix3, bn_pix3, pixel, nullptr, pix_out, 32768, 256, 128);
}
